// VectorQuantizerEMA_58523224376026
// MI455X (gfx1250) — compile-verified, matrix operand order checked by reading, not run on hardware
//
#include <hip/hip_runtime.h>
#include <hip/hip_bf16.h>

typedef __attribute__((ext_vector_type(16))) __bf16 v16bf;
typedef __attribute__((ext_vector_type(8)))  float  v8f;

#define VQ_N 131072
#define VQ_D 128
#define VQ_K 1024
#define ROWS_PER_BLOCK 32      // M tile (two 16-row WMMA tiles)
#define CODE_TILES (VQ_K / 16) // 64 code tiles of 16 codes
#define WAVES_PER_BLOCK 4
#define TILES_PER_WAVE (CODE_TILES / WAVES_PER_BLOCK) // 16
// frag element stride per code tile: 4 frags * 32 lanes * 16 bf16
#define FRAG_TILE_STRIDE 2048
#define FRAG_F_STRIDE    512

// fp32 -> bf16 round-to-nearest-even
static __device__ inline unsigned short f2bf(float x) {
    union { float f; unsigned u; } c; c.f = x;
    unsigned u = c.u;
    return (unsigned short)((u + 0x7FFFu + ((u >> 16) & 1u)) >> 16);
}

union BF16Frag {
    v16bf v;
    unsigned short us[16];
    unsigned u32[8];
};

static __device__ inline void load_bfrag(BF16Frag* b, const unsigned short* p) {
#pragma unroll
    for (int f = 0; f < 4; ++f)
        b[f].v = *(const v16bf*)(p + f * FRAG_F_STRIDE);
}

// Pin packed fragment dwords in VGPRs; makes their def opaque so the
// compiler cannot rematerialize the f32->bf16 packing inside the k-loop.
static __device__ inline void pin_frag(BF16Frag* a) {
#pragma unroll
    for (int f = 0; f < 4; ++f)
#pragma unroll
        for (int i = 0; i < 8; ++i)
            asm volatile("" : "+v"(a[f].u32[i]));
}

// ---------------------------------------------------------------------------
// Prep: swizzle codebook into WMMA-B bf16 fragments in ws, compute ||e_k||^2,
// zero the loss accumulator. One block per 16-code tile.
// frag layout: element ((c*4 + f)*32 + lane)*16 + (v*2 + h), bf16.
// ---------------------------------------------------------------------------
__global__ void vq_prep_kernel(const float* __restrict__ E,
                               unsigned short* __restrict__ frag,
                               float* __restrict__ enorm,
                               float* __restrict__ loss_slot) {
    const int tid  = threadIdx.x;      // 0..127
    const int lane = tid & 31;
    const int f    = tid >> 5;         // feature chunk 0..3
    const int c    = blockIdx.x;       // code tile 0..63

    const int ncol = lane & 15;
    const int grp  = lane >> 4;
    const int code = c * 16 + ncol;

    unsigned short* dst = frag + (((size_t)(c * 4 + f) * 32 + lane) * 16);
#pragma unroll
    for (int v = 0; v < 8; ++v) {
#pragma unroll
        for (int h = 0; h < 2; ++h) {
            const int k = ((v & 3) * 2 + h) + grp * 8 + (v >> 2) * 16;
            dst[v * 2 + h] = f2bf(E[(size_t)code * VQ_D + f * 32 + k]);
        }
    }

    if (tid < 16) {
        const int ck = c * 16 + tid;
        float s = 0.f;
#pragma unroll 8
        for (int d = 0; d < VQ_D; ++d) {
            const float e = E[(size_t)ck * VQ_D + d];
            s = fmaf(e, e, s);
        }
        enorm[ck] = s;
    }
    if (blockIdx.x == 0 && tid == 0) *loss_slot = 0.f;
}

// ---------------------------------------------------------------------------
// Main: one block = 32 input rows vs all 1024 codes.
// 4 waves; wave w scans 16 code tiles with double-buffered B fragments and
// two interleaved accumulator chains (rows 0-15 and 16-31).
// ---------------------------------------------------------------------------
__global__ void __launch_bounds__(128)
vq_main_kernel(const float* __restrict__ X,
               const float* __restrict__ E,
               const unsigned short* __restrict__ frag,
               const float* __restrict__ enorm,
               float* __restrict__ out_q,
               float* __restrict__ out_loss,
               float* __restrict__ out_idx) {
    __shared__ float xtile[ROWS_PER_BLOCK * VQ_D];     // 16 KB
    __shared__ float senorm[VQ_K];                     // 4 KB
    __shared__ float redval[ROWS_PER_BLOCK * 64];      // 8 KB
    __shared__ int   redidx[ROWS_PER_BLOCK * 64];      // 8 KB
    __shared__ int   winidx[ROWS_PER_BLOCK];
    __shared__ float rbuf[128];

    const int tid  = threadIdx.x;     // 0..127
    const int lane = tid & 31;
    const int wave = tid >> 5;
    const int grp  = lane >> 4;
    const int mrow = lane & 15;
    const size_t nbase = (size_t)blockIdx.x * ROWS_PER_BLOCK;

    // Stage X tile (32 x 128 fp32) and code norms into LDS
    {
        const float4* Xv = (const float4*)(X + nbase * VQ_D);
        float4* xt4 = (float4*)xtile;
        for (int i = tid; i < ROWS_PER_BLOCK * VQ_D / 4; i += 128)
            xt4[i] = Xv[i];
        for (int i = tid; i < VQ_K; i += 128)
            senorm[i] = enorm[i];
    }
    __syncthreads();

    // Build two A-matrix bf16 fragment sets (rows 0-15 and rows 16-31)
    BF16Frag a0[4], a1[4];
#pragma unroll
    for (int f = 0; f < 4; ++f) {
#pragma unroll
        for (int v = 0; v < 8; ++v) {
#pragma unroll
            for (int h = 0; h < 2; ++h) {
                const int k = ((v & 3) * 2 + h) + grp * 8 + (v >> 2) * 16;
                a0[f].us[v * 2 + h] = f2bf(xtile[mrow * VQ_D + f * 32 + k]);
                a1[f].us[v * 2 + h] = f2bf(xtile[(16 + mrow) * VQ_D + f * 32 + k]);
            }
        }
    }
    // Keep only the packed form live across the k-loop (no remat of packing)
    pin_frag(a0);
    pin_frag(a1);

    float bval0[8], bval1[8];
    int   bidx0[8], bidx1[8];
#pragma unroll
    for (int v = 0; v < 8; ++v) {
        bval0[v] = 3.4e38f; bidx0[v] = 0;
        bval1[v] = 3.4e38f; bidx1[v] = 0;
    }

    const unsigned short* bp =
        frag + (size_t)wave * TILES_PER_WAVE * FRAG_TILE_STRIDE + (size_t)lane * 16;

    auto process = [&](int i, const BF16Frag* b) {
        v8f acc0 = {}, acc1 = {};
#pragma unroll
        for (int f = 0; f < 4; ++f) {
            acc0 = __builtin_amdgcn_wmma_f32_16x16x32_bf16(
                false, a0[f].v, false, b[f].v, (short)0, acc0, false, false);
            acc1 = __builtin_amdgcn_wmma_f32_16x16x32_bf16(
                false, a1[f].v, false, b[f].v, (short)0, acc1, false, false);
        }
        const int   c    = wave * TILES_PER_WAVE + i;
        const int   code = c * 16 + mrow;
        const float en   = senorm[code];
#pragma unroll
        for (int v = 0; v < 8; ++v) {
            const float d0 = fmaf(-2.f, acc0[v], en);  // ||e||^2 - 2 x.e
            if (d0 < bval0[v]) { bval0[v] = d0; bidx0[v] = code; }
            const float d1 = fmaf(-2.f, acc1[v], en);
            if (d1 < bval1[v]) { bval1[v] = d1; bidx1[v] = code; }
        }
    };

    // Double-buffered scan over this wave's 16 code tiles
    BF16Frag b0[4], b1[4];
    load_bfrag(b0, bp);
    for (int i = 0; i < TILES_PER_WAVE; i += 2) {
        load_bfrag(b1, bp + FRAG_TILE_STRIDE);
        process(i, b0);
        if (i + 2 < TILES_PER_WAVE)
            load_bfrag(b0, bp + 2 * FRAG_TILE_STRIDE);
        process(i + 1, b1);
        bp += 2 * FRAG_TILE_STRIDE;
    }

    // Scatter per-lane candidates; C/D layout: row = v + 8*grp (+16 for acc1)
#pragma unroll
    for (int v = 0; v < 8; ++v) {
        const int row = v + 8 * grp;
        const int col = wave * 16 + mrow;
        redval[row * 64 + col]        = bval0[v];
        redidx[row * 64 + col]        = bidx0[v];
        redval[(16 + row) * 64 + col] = bval1[v];
        redidx[(16 + row) * 64 + col] = bidx1[v];
    }
    __syncthreads();

    // One thread per row: argmin over 64 candidates (tie -> smaller index)
    if (tid < ROWS_PER_BLOCK) {
        float bv = 3.4e38f; int bi = 0;
        for (int j = 0; j < 64; ++j) {
            const float v  = redval[tid * 64 + j];
            const int   ix = redidx[tid * 64 + j];
            if (v < bv || (v == bv && ix < bi)) { bv = v; bi = ix; }
        }
        winidx[tid] = bi;
        out_idx[nbase + tid] = (float)bi;
    }
    __syncthreads();

    // Gather winning codebook rows (fp32), write quantized, fuse loss
    float lsum = 0.f;
#pragma unroll 4
    for (int m = 0; m < ROWS_PER_BLOCK; ++m) {
        const int idx = winidx[m];
        const float q = E[(size_t)idx * VQ_D + tid];
        const float x = xtile[m * VQ_D + tid];
        out_q[(nbase + m) * VQ_D + tid] = q;
        const float d = q - x;
        lsum = fmaf(d, d, lsum);
    }

    rbuf[tid] = lsum;
    __syncthreads();
#pragma unroll
    for (int s = 64; s > 0; s >>= 1) {
        if (tid < s) rbuf[tid] += rbuf[tid + s];
        __syncthreads();
    }
    if (tid == 0) {
        // loss = q_latent + 0.5*e_latent; both equal MSE numerically
        const float scale = 1.5f / ((float)VQ_N * (float)VQ_D);
        atomicAdd(out_loss, rbuf[0] * scale);
    }
}

extern "C" void kernel_launch(void* const* d_in, const int* in_sizes, int n_in,
                              void* d_out, int out_size, void* d_ws, size_t ws_size,
                              hipStream_t stream) {
    const float* X = (const float*)d_in[0];   // [N, D] fp32
    const float* E = (const float*)d_in[1];   // [K, D] fp32

    float* out      = (float*)d_out;
    float* out_q    = out;                                   // N*D
    float* out_loss = out + (size_t)VQ_N * VQ_D;             // 1
    float* out_idx  = out + (size_t)VQ_N * VQ_D + 1;         // N (as float)

    unsigned short* frag = (unsigned short*)d_ws;            // 256 KB
    float* enorm = (float*)((char*)d_ws + (size_t)CODE_TILES * FRAG_TILE_STRIDE * 2);

    vq_prep_kernel<<<CODE_TILES, 128, 0, stream>>>(E, frag, enorm, out_loss);
    vq_main_kernel<<<VQ_N / ROWS_PER_BLOCK, 128, 0, stream>>>(
        X, E, frag, enorm, out_q, out_loss, out_idx);
}
